// MultichannelMultiheadAttention_10840497455860
// MI455X (gfx1250) — matrix operand order repaired and checked by reading, not compiled, not run on hardware
//
#include <hip/hip_runtime.h>
#include <hip/hip_bf16.h>

#define B_  2
#define C_  4
#define F_  512
#define H_  8
#define W_  1024
#define KC  3
#define DH  64
#define BC  (B_*C_)

typedef __bf16 bf16;
typedef __attribute__((ext_vector_type(16))) __bf16 v16bf;
typedef __attribute__((ext_vector_type(8)))  __bf16 v8bf;
typedef __attribute__((ext_vector_type(2)))  __bf16 v2bf;
typedef __attribute__((ext_vector_type(8)))  float  v8f;
typedef __attribute__((ext_vector_type(4)))  float  v4f;

union AFrag { v16bf v; v8bf h[2]; };

__device__ __forceinline__ v8f wmma_bf16(const AFrag& a, const AFrag& b, v8f c) {
    return __builtin_amdgcn_wmma_f32_16x16x32_bf16(
        false, a.v, false, b.v, (short)0, c, false, false);
}

__device__ __forceinline__ void wait_async0() {
#if __has_builtin(__builtin_amdgcn_s_wait_asynccnt)
    __builtin_amdgcn_s_wait_asynccnt(0);
#else
    asm volatile("s_wait_asynccnt 0" ::: "memory");
#endif
}

// ---------------------------------------------------------------------------
// GEMM: Y[bc][g][w] = sum_f Wl[c][g][f] * X[bc][f][w]   (M=512, N=1024, K=512)
// tile 128x128 per WG, 8 waves (2x4), each wave 64x32 = 4x2 C-tiles, K-step 32
// Staging loads are hoisted into register arrays so they can issue as a clause.
// ---------------------------------------------------------------------------
#define GPAD 40   // padded LDS row (bf16), 80B = 16B-aligned multiples

__global__ __launch_bounds__(256)
void gemm_wl(const float* __restrict__ Wl,   // [C][F][F]
             const float* __restrict__ X,    // [BC][F][W]
             float* __restrict__ Y)          // [BC][F][W]
{
    __shared__ bf16 Asm[128 * GPAD];
    __shared__ bf16 Bsm[128 * GPAD];

    const int t    = threadIdx.x;
    const int lane = t & 31, wid = t >> 5;
    const int wm   = wid >> 2, wn = wid & 3;
    const int half = lane >> 4, l16 = lane & 15;
    const int kb   = half * 8;

    const int n0 = blockIdx.x * 128;
    const int m0 = blockIdx.y * 128;
    const int bc = blockIdx.z;
    const int c  = bc % C_;

    const float* A  = Wl + (size_t)c  * F_ * F_;
    const float* Xb = X  + (size_t)bc * F_ * W_;
    float*       Yb = Y  + (size_t)bc * F_ * W_;

    v8f acc[4][2] = {};

    for (int k0 = 0; k0 < F_; k0 += 32) {
        // ---- stage all global loads first (independent regs -> load clause)
        v4f a4[4];
        #pragma unroll
        for (int i = 0; i < 4; ++i) {
            int idx = t + i * 256;                 // 0..1023
            int r = idx >> 3, c4 = (idx & 7) << 2;
            a4[i] = *(const v4f*)(A + (size_t)(m0 + r) * F_ + k0 + c4);
        }
        v4f xb[2][2];
        #pragma unroll
        for (int i = 0; i < 2; ++i) {
            int idx = t + i * 256;                 // 0..511
            int kp2 = idx >> 5;                    // k-pair index 0..15
            int n4  = (idx & 31) << 2;             // 0..124
            const float* xr = Xb + (size_t)(k0 + 2 * kp2) * W_ + n0 + n4;
            xb[i][0] = *(const v4f*)(xr);
            xb[i][1] = *(const v4f*)(xr + W_);
        }
        // ---- convert + packed b32 LDS stores
        #pragma unroll
        for (int i = 0; i < 4; ++i) {
            int idx = t + i * 256;
            int r = idx >> 3, c4 = (idx & 7) << 2;
            v2bf p0; p0.x = (bf16)a4[i].x; p0.y = (bf16)a4[i].y;
            v2bf p1; p1.x = (bf16)a4[i].z; p1.y = (bf16)a4[i].w;
            *(v2bf*)&Asm[r * GPAD + c4]     = p0;
            *(v2bf*)&Asm[r * GPAD + c4 + 2] = p1;
        }
        #pragma unroll
        for (int i = 0; i < 2; ++i) {
            int idx = t + i * 256;
            int kp2 = idx >> 5;
            int n4  = (idx & 31) << 2;
            #pragma unroll
            for (int j = 0; j < 4; ++j) {
                v2bf p; p.x = (bf16)xb[i][0][j]; p.y = (bf16)xb[i][1][j];
                *(v2bf*)&Bsm[(n4 + j) * GPAD + 2 * kp2] = p;
            }
        }
        __syncthreads();

        AFrag bfr[2];
        #pragma unroll
        for (int nt = 0; nt < 2; ++nt) {
            const bf16* bp = &Bsm[(wn * 32 + nt * 16 + l16) * GPAD];
            bfr[nt].h[0] = *(const v8bf*)(bp + kb);
            bfr[nt].h[1] = *(const v8bf*)(bp + 16 + kb);
        }
        #pragma unroll
        for (int mt = 0; mt < 4; ++mt) {
            AFrag af;
            const bf16* ap = &Asm[(wm * 64 + mt * 16 + l16) * GPAD];
            af.h[0] = *(const v8bf*)(ap + kb);
            af.h[1] = *(const v8bf*)(ap + 16 + kb);
            #pragma unroll
            for (int nt = 0; nt < 2; ++nt)
                acc[mt][nt] = wmma_bf16(af, bfr[nt], acc[mt][nt]);
        }
        __syncthreads();
    }

    #pragma unroll
    for (int mt = 0; mt < 4; ++mt)
        #pragma unroll
        for (int nt = 0; nt < 2; ++nt) {
            int gm = m0 + wm * 64 + mt * 16 + half * 8;
            int gn = n0 + wn * 32 + nt * 16 + l16;
            #pragma unroll
            for (int r = 0; r < 8; ++r)
                Yb[(size_t)(gm + r) * W_ + gn] = acc[mt][nt][r];
        }
}

// ---------------------------------------------------------------------------
// 1x3 channel-mix conv + bias + head split + optional rotary.
// one thread per (bc, h, w, d-pair); w fastest for coalesced reads of lin.
// ---------------------------------------------------------------------------
__global__ __launch_bounds__(256)
void conv_rope(const float* __restrict__ lin,   // [BC][F][W]
               const float* __restrict__ wcv,   // [C][C][KC]
               const float* __restrict__ bias,  // [C]
               bf16* __restrict__ out,          // [BC][H][W][DH]
               int do_rope)
{
    int g  = blockIdx.x * 256 + threadIdx.x;   // BC*H*32*W = 2^21 threads
    int w  = g & (W_ - 1);
    int dp = (g >> 10) & 31;
    int h  = (g >> 15) & 7;
    int bc = g >> 18;
    int b  = bc >> 2, o = bc & 3;
    int f0 = h * DH + 2 * dp;

    float y0 = bias[o], y1 = bias[o];
    #pragma unroll
    for (int i = 0; i < C_; ++i) {
        const float* lp0 = lin + ((size_t)(b * C_ + i) * F_ + f0) * W_;
        const float* lp1 = lp0 + W_;
        const float* wp  = wcv + (o * C_ + i) * KC;
        #pragma unroll
        for (int tt = 0; tt < KC; ++tt) {
            int wx = w + tt - 1;
            if (wx >= 0 && wx < W_) {
                y0 += lp0[wx] * wp[tt];
                y1 += lp1[wx] * wp[tt];
            }
        }
    }
    if (do_rope) {
        // theta = w / 10000^{(2*dp)/DH}
        float inv = __expf(-((float)(2 * dp) / (float)DH) * 9.210340371976184f);
        float th  = (float)w * inv;
        float sn, cs;
        __sincosf(th, &sn, &cs);
        float r0 = y0 * cs - y1 * sn;
        float r1 = y1 * cs + y0 * sn;
        y0 = r0; y1 = r1;
    }
    v2bf p; p.x = (bf16)y0; p.y = (bf16)y1;
    *(v2bf*)(out + (((size_t)bc * H_ + h) * W_ + w) * DH + 2 * dp) = p;
}

// ---------------------------------------------------------------------------
// Flash attention per (b,c,h): WG handles 128 q rows (8 waves x 16 rows).
// Per 64-kpos tile: K tile staged via global_load_async_to_lds_b128 (ASYNCcnt),
// S = Q K^T (WMMA), add scale+prev+mask, emit qk logits (non-temporal stores;
// prev_qk read non-temporal so the streams don't evict mask/Q/K/V from L2),
// online softmax (16-lane shuffles), P -> LDS -> A-frag, acc += P V (WMMA).
// ---------------------------------------------------------------------------
#define VPAD 72

__global__ __launch_bounds__(256)
void attn(const bf16* __restrict__ Q,     // [BC*H][W][DH]
          const bf16* __restrict__ Kk,
          const bf16* __restrict__ V,
          const float* __restrict__ prev, // [BC*H][W][W]
          const float* __restrict__ mask, // [W][W]
          float* __restrict__ qkout,      // [BC*H][W][W]
          float* __restrict__ aout)       // [BC][F][W]
{
    __shared__ bf16 Ksm[64 * VPAD];        // K tile  [kpos][d]  (row major)
    __shared__ bf16 Vts[DH * VPAD];        // V tile  [d][kpos]  (transposed)
    __shared__ bf16 Psm[8 * 16 * VPAD];    // per-wave P tile [16 q][64 kpos]

    const int t    = threadIdx.x;
    const int lane = t & 31, wid = t >> 5;
    const int half = lane >> 4, l16 = lane & 15;
    const int kb   = half * 8;

    const int q0  = blockIdx.x * 128;
    const int bch = blockIdx.y;            // bc*H + h
    const int h   = bch & 7;
    const int bc  = bch >> 3;

    const bf16* Qh = Q  + (size_t)bch * W_ * DH;
    const bf16* Kh = Kk + (size_t)bch * W_ * DH;
    const bf16* Vh = V  + (size_t)bch * W_ * DH;
    const float* ph = prev + (size_t)bch * W_ * W_;
    float* qko = qkout + (size_t)bch * W_ * W_;

    const int qw = q0 + wid * 16;          // this wave's 16 q rows

    AFrag qf[2];
    {
        const bf16* qp = Qh + (size_t)(qw + l16) * DH;
        qf[0].h[0] = *(const v8bf*)(qp + kb);
        qf[0].h[1] = *(const v8bf*)(qp + 16 + kb);
        qf[1].h[0] = *(const v8bf*)(qp + 32 + kb);
        qf[1].h[1] = *(const v8bf*)(qp + 48 + kb);
    }

    v8f acc[4] = {};
    float mrun[8], lrun[8];
    #pragma unroll
    for (int r = 0; r < 8; ++r) { mrun[r] = -1e30f; lrun[r] = 0.f; }
    const float sc = 0.044194173824159216f;   // 1/sqrt(512)

    // precomputed loop-invariant per-lane pieces for the async K copy
    const int akk  = t >> 2;               // 0..63 (k row)
    const int aseg = (t & 3) << 4;         // 0,16,32,48 (d offset, elements)

    for (int kt = 0; kt < W_; kt += 64) {
        // --- K tile: async global->LDS copy (2 x 16B per thread) ---
        {
            unsigned long long g0 =
                (unsigned long long)(const void*)(Kh + (size_t)(kt + akk) * DH + aseg);
            unsigned l0 = (unsigned)(unsigned long long)(const void*)&Ksm[akk * VPAD + aseg];
            asm volatile("global_load_async_to_lds_b128 %0, %1, off"
                         :: "v"(l0), "v"(g0) : "memory");
            asm volatile("global_load_async_to_lds_b128 %0, %1, off offset:16"
                         :: "v"(l0), "v"(g0) : "memory");
        }
        // --- V tile transpose: k-pairs packed into b32 LDS stores ---
        {
            int kk2 = t >> 3;              // 0..31 (k pair)
            int d0  = (t & 7) << 3;        // 0..56
            const bf16* vp = Vh + (size_t)(kt + 2 * kk2) * DH + d0;
            v8bf v0 = *(const v8bf*)(vp);
            v8bf v1 = *(const v8bf*)(vp + DH);
            #pragma unroll
            for (int j = 0; j < 8; ++j) {
                v2bf p; p.x = v0[j]; p.y = v1[j];
                *(v2bf*)&Vts[(d0 + j) * VPAD + 2 * kk2] = p;
            }
        }
        wait_async0();
        __syncthreads();

        // logits: 4 n-tiles of 16x16
        v8f S[4];
        #pragma unroll
        for (int nt = 0; nt < 4; ++nt) {
            AFrag bf0, bf1;
            const bf16* bp = &Ksm[(nt * 16 + l16) * VPAD];
            bf0.h[0] = *(const v8bf*)(bp + kb);
            bf0.h[1] = *(const v8bf*)(bp + 16 + kb);
            bf1.h[0] = *(const v8bf*)(bp + 32 + kb);
            bf1.h[1] = *(const v8bf*)(bp + 48 + kb);
            v8f s = {};
            s = wmma_bf16(qf[0], bf0, s);
            s = wmma_bf16(qf[1], bf1, s);

            // scale + prev(NT) + mask(RT), emit qk (NT), keep for softmax
            int kcol = kt + nt * 16 + l16;
            const float* pp = ph   + (size_t)(qw + half * 8) * W_ + kcol;
            const float* mp = mask + (size_t)(qw + half * 8) * W_ + kcol;
            float*       qo = qko  + (size_t)(qw + half * 8) * W_ + kcol;
            #pragma unroll
            for (int r = 0; r < 8; ++r) {
                float pv  = __builtin_nontemporal_load(pp + (size_t)r * W_);
                float val = s[r] * sc + pv + mp[(size_t)r * W_];
                __builtin_nontemporal_store(val, qo + (size_t)r * W_);
                s[r] = val;
            }
            S[nt] = s;
        }

        // online softmax over this 64-wide block
        #pragma unroll
        for (int r = 0; r < 8; ++r) {
            float m = fmaxf(fmaxf(S[0][r], S[1][r]), fmaxf(S[2][r], S[3][r]));
            #pragma unroll
            for (int d = 1; d < 16; d <<= 1) m = fmaxf(m, __shfl_xor(m, d, 32));
            float mn  = fmaxf(mrun[r], m);
            float scl = __expf(mrun[r] - mn);
            lrun[r] *= scl;
            #pragma unroll
            for (int dt = 0; dt < 4; ++dt) acc[dt][r] *= scl;
            float rs = 0.f;
            #pragma unroll
            for (int nt = 0; nt < 4; ++nt) {
                float p = __expf(S[nt][r] - mn);
                S[nt][r] = p;
                rs += p;
            }
            #pragma unroll
            for (int d = 1; d < 16; d <<= 1) rs += __shfl_xor(rs, d, 32);
            lrun[r] += rs;
            mrun[r] = mn;
        }

        // P (C layout) -> LDS -> A-frag layout; per-wave region, in-order LDS
        bf16* pw = &Psm[wid * 16 * VPAD];
        #pragma unroll
        for (int nt = 0; nt < 4; ++nt)
            #pragma unroll
            for (int r = 0; r < 8; ++r)
                pw[(r + 8 * half) * VPAD + nt * 16 + l16] = (bf16)S[nt][r];

        AFrag pf[2];
        {
            const bf16* pp2 = pw + l16 * VPAD;
            pf[0].h[0] = *(const v8bf*)(pp2 + kb);
            pf[0].h[1] = *(const v8bf*)(pp2 + 16 + kb);
            pf[1].h[0] = *(const v8bf*)(pp2 + 32 + kb);
            pf[1].h[1] = *(const v8bf*)(pp2 + 48 + kb);
        }
        #pragma unroll
        for (int dt = 0; dt < 4; ++dt) {
            AFrag vf0, vf1;
            const bf16* vp = &Vts[(dt * 16 + l16) * VPAD];
            vf0.h[0] = *(const v8bf*)(vp + kb);
            vf0.h[1] = *(const v8bf*)(vp + 16 + kb);
            vf1.h[0] = *(const v8bf*)(vp + 32 + kb);
            vf1.h[1] = *(const v8bf*)(vp + 48 + kb);
            acc[dt] = wmma_bf16(pf[0], vf0, acc[dt]);
            acc[dt] = wmma_bf16(pf[1], vf1, acc[dt]);
        }
        __syncthreads();
    }

    // normalize, write a in [bc][f = h*64+d][w = q] layout
    float* ao = aout + ((size_t)bc * F_ + h * DH) * W_;
    #pragma unroll
    for (int dt = 0; dt < 4; ++dt) {
        int d = dt * 16 + l16;
        #pragma unroll
        for (int r = 0; r < 8; ++r) {
            int q = qw + 8 * half + r;
            ao[(size_t)d * W_ + q] = acc[dt][r] / lrun[r];
        }
    }
}

// ---------------------------------------------------------------------------
extern "C" void kernel_launch(void* const* d_in, const int* in_sizes, int n_in,
                              void* d_out, int out_size, void* d_ws, size_t ws_size,
                              hipStream_t stream) {
    const float* x       = (const float*)d_in[0];
    const float* prev    = (const float*)d_in[1];
    const float* mask    = (const float*)d_in[2];
    const float* wq_lin  = (const float*)d_in[3];
    const float* wq_conv = (const float*)d_in[4];
    const float* bq      = (const float*)d_in[5];
    const float* wk_lin  = (const float*)d_in[6];
    const float* wk_conv = (const float*)d_in[7];
    const float* bk      = (const float*)d_in[8];
    const float* wv_lin  = (const float*)d_in[9];
    const float* wv_conv = (const float*)d_in[10];
    const float* bv      = (const float*)d_in[11];
    const float* wo_lin  = (const float*)d_in[12];

    float* out   = (float*)d_out;                       // [BC][F][W]
    float* qkout = out + (size_t)BC * F_ * W_;          // [BC*H][W][W]

    char* ws   = (char*)d_ws;
    float* lin = (float*)ws;                            // 16 MB fp32 (reused as 'a')
    bf16*  qb  = (bf16*)(ws + (16u << 20));             // 8 MB
    bf16*  kbf = (bf16*)(ws + (24u << 20));             // 8 MB
    bf16*  vb  = (bf16*)(ws + (32u << 20));             // 8 MB

    dim3 gg(W_ / 128, F_ / 128, BC);                    // 8 x 4 x 8
    const int convBlocks = (BC * H_ * 32 * W_) / 256;   // 8192

    gemm_wl<<<gg, 256, 0, stream>>>(wq_lin, x, lin);
    conv_rope<<<convBlocks, 256, 0, stream>>>(lin, wq_conv, bq, qb, 1);
    gemm_wl<<<gg, 256, 0, stream>>>(wk_lin, x, lin);
    conv_rope<<<convBlocks, 256, 0, stream>>>(lin, wk_conv, bk, kbf, 1);
    gemm_wl<<<gg, 256, 0, stream>>>(wv_lin, x, lin);
    conv_rope<<<convBlocks, 256, 0, stream>>>(lin, wv_conv, bv, vb, 0);

    attn<<<dim3(W_ / 128, BC * H_), 256, 0, stream>>>(qb, kbf, vb, prev, mask,
                                                      qkout, lin);
    gemm_wl<<<gg, 256, 0, stream>>>(wo_lin, lin, out);
}
